// _FastARAttention_76811195122133
// MI455X (gfx1250) — matrix operand order repaired and checked by reading, not compiled, not run on hardware
//
#include <hip/hip_runtime.h>
#include <hip/hip_bf16.h>

// ---------------- problem constants (from reference) ----------------
#define B_   2
#define S_   2048
#define H_   2048
#define NH_  16
#define NKV_ 8
#define D_   128
#define OQKV 4096              // NH*D + 2*NKV*D
#define SCALE_ 0.08838834764831843f   // D^-0.5
#define EPS_ 1e-6f
#define NEG_INF_ (-1e30f)

typedef __attribute__((ext_vector_type(16))) __bf16 v16bf;
typedef __attribute__((ext_vector_type(8)))  float  v8f;

union FragAB {
    v16bf v;
    uint4 q[2];
    unsigned int u[8];
    unsigned short h[16];
};

// ---------------- bf16 helpers (round-to-nearest-even) ----------------
__device__ __forceinline__ unsigned short f2bf(float x) {
    unsigned int u = __float_as_uint(x);
    u += 0x7FFFu + ((u >> 16) & 1u);
    return (unsigned short)(u >> 16);
}
__device__ __forceinline__ unsigned int pk2bf(float lo, float hi) {
    return (unsigned int)f2bf(lo) | ((unsigned int)f2bf(hi) << 16);
}
__device__ __forceinline__ float bf2f(unsigned int u16) {
    return __uint_as_float(u16 << 16);
}

// ---------------- WMMA fragment loaders ----------------
// A (16x32, bf16): lane m=l&15; VGPR0-7 hold K = (l>>4)*16 + 0..15.
__device__ __forceinline__ FragAB load_a_bf16(const unsigned short* base, int ld,
                                              int k0, int lane) {
    FragAB f;
    int m = lane & 15, ko = (lane >> 4) * 16;
    const unsigned short* p = base + (size_t)m * ld + k0 + ko;
    f.q[0] = *(const uint4*)(p);
    f.q[1] = *(const uint4*)(p + 8);
    return f;
}
// B (32x16, bf16) built from "N rows x K-contiguous" storage (i.e. B = W^T):
// lane n=l&15 picks the row; lane-half picks K block of 16; VGPR v holds K=blk+2v,2v+1.
__device__ __forceinline__ FragAB load_b_bf16(const unsigned short* base, int ld,
                                              int k0, int lane) {
    FragAB f;
    int n = lane & 15, kb = (lane >> 4) * 16;
    const unsigned short* p = base + (size_t)n * ld + k0 + kb;
    f.q[0] = *(const uint4*)(p);
    f.q[1] = *(const uint4*)(p + 8);
    return f;
}

__device__ __forceinline__ v8f wmma_bf16(const FragAB& a, const FragAB& b, v8f c) {
    return __builtin_amdgcn_wmma_f32_16x16x32_bf16(
        /*neg_a=*/false, a.v, /*neg_b=*/false, b.v,
        /*c_mod=*/(short)0, c, /*reuse_a=*/false, /*reuse_b=*/false);
}

// 16-byte async global->LDS copy (CDNA5, ASYNCcnt-tracked). GV mode.
// Low 32 bits of a generic pointer to a __shared__ object are the LDS byte
// address (generic LDS pointers are {SHARED_BASE, lds_offset}).
__device__ __forceinline__ void async_copy16(const void* lds_dst, const void* gsrc) {
    unsigned lds_off = (unsigned)(uintptr_t)lds_dst;
    unsigned long long ga = (unsigned long long)(uintptr_t)gsrc;
    asm volatile("global_load_async_to_lds_b128 %0, %1, off"
                 :: "v"(lds_off), "v"(ga) : "memory");
}

// =====================================================================
// Kernel 0: bulk fp32 -> bf16 conversion (one pass; amortized over reuse)
// =====================================================================
__global__ __launch_bounds__(256)
void cvt_f32_to_bf16(const float* __restrict__ src, unsigned short* __restrict__ dst,
                     int n4) {
    int i = blockIdx.x * 256 + threadIdx.x;
    if (i < n4) {
        float4 v = ((const float4*)src)[i];
        ((uint2*)dst)[i] = make_uint2(pk2bf(v.x, v.y), pk2bf(v.z, v.w));
    }
}

// =====================================================================
// Kernel 1/4: C[M,N] = A[M,K] * W[N,K]^T  (all-bf16 WMMA, fp32 accumulate)
// block = 128 threads = 4 waves; tile 64(M) x 64(N).
// W tiles double-buffered into LDS with global_load_async_to_lds_b128.
// =====================================================================
template<bool OUT_F32>
__global__ __launch_bounds__(128)
void wmma_gemm_nt(const unsigned short* __restrict__ A,
                  const unsigned short* __restrict__ Wb,
                  void* __restrict__ Cp, int M, int N, int K) {
    __shared__ alignas(16) unsigned short lw[2][64][40];  // 64 N-rows x 32 K, padded
    const int t    = threadIdx.x;
    const int lane = t & 31;
    const int wave = t >> 5;
    const int n0   = blockIdx.x * 64;
    const int m0   = blockIdx.y * 64 + wave * 16;

    v8f acc[4];
#pragma unroll
    for (int i = 0; i < 4; i++) acc[i] = {};

    // stage one 64x32 bf16 W tile: 4 KB = 256 x 16B chunks, 2 per thread
    auto stage_w = [&](int k0, int buf) {
#pragma unroll
        for (int j = 0; j < 2; j++) {
            int chunk = t + 128 * j;
            int row = chunk >> 2, part = chunk & 3;
            async_copy16(&lw[buf][row][part * 8],
                         Wb + (size_t)(n0 + row) * K + k0 + part * 8);
        }
    };

    stage_w(0, 0);
    for (int k0 = 0; k0 < K; k0 += 32) {
        const int buf = (k0 >> 5) & 1;
        const bool havenext = (k0 + 32 < K);
        if (havenext) {
            stage_w(k0 + 32, buf ^ 1);
            asm volatile("s_wait_asynccnt 0x2" ::: "memory"); // current tile landed
        } else {
            asm volatile("s_wait_asynccnt 0x0" ::: "memory");
        }
        __syncthreads();

        FragAB af = load_a_bf16(A + (size_t)m0 * K, K, k0, lane);
#pragma unroll
        for (int nt = 0; nt < 4; nt++) {
            FragAB bf = load_b_bf16(&lw[buf][nt * 16][0], 40, 0, lane);
            acc[nt] = wmma_bf16(af, bf, acc[nt]);
        }
        __syncthreads();   // done reading buf before it is re-staged
    }

    // epilogue: C layout -> lane n=l&15, VGPR r -> row r + (l>>4)*8
#pragma unroll
    for (int nt = 0; nt < 4; nt++) {
#pragma unroll
        for (int r = 0; r < 8; r++) {
            int gm = m0 + r + ((lane >> 4) * 8);
            int gn = n0 + nt * 16 + (lane & 15);
            float v = acc[nt][r];
            if (OUT_F32) ((float*)Cp)[(size_t)gm * N + gn] = v;
            else ((unsigned short*)Cp)[(size_t)gm * N + gn] = f2bf(v);
        }
    }
}

// =====================================================================
// Kernel 2: per-(b,s,head) RMSNorm + GPT-J RoPE + repack.
// One wave per 128-wide row. hh: 0..15 -> Q head, 16..23 -> K head, 24..31 -> V head.
// Q -> [b,h,s,d] bf16 ; K -> [b,kvh,s,d] bf16 ; V -> transposed Vt[b,kvh,d,s] bf16.
// =====================================================================
__global__ __launch_bounds__(128)
void norm_rope_pack(const unsigned short* __restrict__ qkv,
                    const long long* __restrict__ pos_ids,
                    const float* __restrict__ qw, const float* __restrict__ kw,
                    unsigned short* __restrict__ Qb,
                    unsigned short* __restrict__ Kb,
                    unsigned short* __restrict__ Vt) {
    const int lane = threadIdx.x & 31;
    const int wave = threadIdx.x >> 5;
    const int row  = blockIdx.x * 4 + wave;       // [0, B*S*32)
    const int b   = row / (S_ * 32);
    const int rem = row % (S_ * 32);
    const int s   = rem / 32;
    const int hh  = rem % 32;

    const unsigned short* src = qkv + ((size_t)(b * S_ + s) * OQKV) + hh * D_ + lane * 4;
    uint2 raw = *(const uint2*)src;
    unsigned int h01 = raw.x, h23 = raw.y;

    if (hh >= 24) {   // V: no norm/rope, just scatter into transposed layout
        int kvh = hh - 24;
        size_t base = ((size_t)(b * NKV_ + kvh) * D_ + lane * 4) * S_ + s;
        Vt[base + 0 * S_] = (unsigned short)(h01 & 0xffffu);
        Vt[base + 1 * S_] = (unsigned short)(h01 >> 16);
        Vt[base + 2 * S_] = (unsigned short)(h23 & 0xffffu);
        Vt[base + 3 * S_] = (unsigned short)(h23 >> 16);
        return;
    }

    float x0 = bf2f(h01 & 0xffffu), x1 = bf2f(h01 >> 16);
    float x2 = bf2f(h23 & 0xffffu), x3 = bf2f(h23 >> 16);

    // RMS over D=128 (32 lanes x 4)
    float ss = x0 * x0 + x1 * x1 + x2 * x2 + x3 * x3;
    ss += __shfl_xor(ss, 1);  ss += __shfl_xor(ss, 2);
    ss += __shfl_xor(ss, 4);  ss += __shfl_xor(ss, 8);
    ss += __shfl_xor(ss, 16);
    float rms = rsqrtf(ss * (1.0f / (float)D_) + EPS_);

    const float* nw = (hh < NH_) ? qw : kw;
    float4 wv = *(const float4*)(nw + lane * 4);
    x0 *= rms * wv.x; x1 *= rms * wv.y; x2 *= rms * wv.z; x3 *= rms * wv.w;

    // GPT-J RoPE: pairs (x[2j], x[2j+1]); lane owns pair indices j0=2*lane, j1=2*lane+1
    float p = (float)pos_ids[(size_t)b * S_ + s];
    const float NEG_LNTH_64 = -9.210340371976184f / 64.0f;   // -ln(10000)/64
    float f0 = p * __expf((float)(2 * lane)     * NEG_LNTH_64);
    float f1 = p * __expf((float)(2 * lane + 1) * NEG_LNTH_64);
    float c0 = cosf(f0), s0 = sinf(f0);
    float c1 = cosf(f1), s1 = sinf(f1);
    float o0 = x0 * c0 - x1 * s0;
    float o1 = x1 * c0 + x0 * s0;
    float o2 = x2 * c1 - x3 * s1;
    float o3 = x3 * c1 + x2 * s1;

    uint2 packed = make_uint2(pk2bf(o0, o1), pk2bf(o2, o3));
    if (hh < NH_) {
        unsigned short* dst = Qb + (((size_t)(b * NH_ + hh) * S_ + s) * D_) + lane * 4;
        *(uint2*)dst = packed;
    } else {
        int kvh = hh - NH_;
        unsigned short* dst = Kb + (((size_t)(b * NKV_ + kvh) * S_ + s) * D_) + lane * 4;
        *(uint2*)dst = packed;
    }
}

// =====================================================================
// Kernel 3: causal GQA flash attention, one wave per 16-row Q tile.
// Key tiles of 32; 8 score-WMMA + 8 PV-WMMA per tile; online softmax in fp32.
// Unmasked/masked tile loops split; softmax scale folded into the exponent.
// =====================================================================
__global__ __launch_bounds__(128)
void flash_attn(const unsigned short* __restrict__ Qb,
                const unsigned short* __restrict__ Kb,
                const unsigned short* __restrict__ Vt,
                unsigned short* __restrict__ AO) {
    __shared__ alignas(16) unsigned short lP[4][16][40];   // wave-private P staging

    const int lane = threadIdx.x & 31;
    const int wave = threadIdx.x >> 5;
    const int bid  = blockIdx.x;                 // [0, B*NH*32)
    const int bh   = bid / 32;
    const int qg   = bid % 32;
    const int b    = bh / NH_;
    const int h    = bh % NH_;
    const int kvh  = h >> 1;                     // G = NH/NKV = 2
    const int q0   = (qg * 4 + wave) * 16;

    const unsigned short* Qbase = Qb + ((size_t)(b * NH_ + h) * S_) * D_;
    const unsigned short* Kbase = Kb + ((size_t)(b * NKV_ + kvh) * S_) * D_;
    const unsigned short* Vbase = Vt + ((size_t)(b * NKV_ + kvh) * D_) * S_;

    // Q fragments for d = 0..127 (4 chunks of 32), kept in registers
    FragAB qf[4];
#pragma unroll
    for (int c = 0; c < 4; c++)
        qf[c] = load_a_bf16(Qbase + (size_t)q0 * D_, D_, 32 * c, lane);

    v8f o[8];
#pragma unroll
    for (int dt = 0; dt < 8; dt++) o[dt] = {};
    float mstat[8], lstat[8];   // running max kept in RAW (unscaled) score units
#pragma unroll
    for (int r = 0; r < 8; r++) { mstat[r] = NEG_INF_; lstat[r] = 0.0f; }

    auto process_tile = [&](int kt0, bool masked) {
        // ---- raw scores S = Q K^T in C layout ----
        float sc[2][8];
#pragma unroll
        for (int nt = 0; nt < 2; nt++) {
            v8f sacc = {};
#pragma unroll
            for (int c = 0; c < 4; c++) {
                FragAB kf = load_b_bf16(Kbase + (size_t)(kt0 + nt * 16) * D_,
                                        D_, 32 * c, lane);
                sacc = wmma_bf16(qf[c], kf, sacc);
            }
            int col = kt0 + nt * 16 + (lane & 15);
#pragma unroll
            for (int r = 0; r < 8; r++) {
                float v = sacc[r];
                if (masked) {
                    int rowq = q0 + r + ((lane >> 4) * 8);
                    v = (col <= rowq) ? v : NEG_INF_;
                }
                sc[nt][r] = v;
            }
        }

        // ---- online softmax (scale folded into the exponent) ----
#pragma unroll
        for (int r = 0; r < 8; r++) {
            float mx = fmaxf(sc[0][r], sc[1][r]);
            mx = fmaxf(mx, __shfl_xor(mx, 1));
            mx = fmaxf(mx, __shfl_xor(mx, 2));
            mx = fmaxf(mx, __shfl_xor(mx, 4));
            mx = fmaxf(mx, __shfl_xor(mx, 8));
            float mnew = fmaxf(mstat[r], mx);
            float fct  = __expf((mstat[r] - mnew) * SCALE_);
            mstat[r] = mnew;
            float p0 = __expf((sc[0][r] - mnew) * SCALE_);
            float p1 = __expf((sc[1][r] - mnew) * SCALE_);
            sc[0][r] = p0; sc[1][r] = p1;
            float ls = p0 + p1;
            ls += __shfl_xor(ls, 1);
            ls += __shfl_xor(ls, 2);
            ls += __shfl_xor(ls, 4);
            ls += __shfl_xor(ls, 8);
            lstat[r] = lstat[r] * fct + ls;
#pragma unroll
            for (int dt = 0; dt < 8; dt++) o[dt][r] *= fct;
        }

        // ---- C-layout -> A-layout for P via wave-private LDS ----
        const int prow0 = (lane >> 4) * 8;
#pragma unroll
        for (int nt = 0; nt < 2; nt++)
#pragma unroll
            for (int r = 0; r < 8; r++)
                lP[wave][r + prow0][nt * 16 + (lane & 15)] = f2bf(sc[nt][r]);
        asm volatile("s_wait_dscnt 0" ::: "memory");

        FragAB pf;
        {
            int m = lane & 15, ko = (lane >> 4) * 16;
            const unsigned short* pp = &lP[wave][m][ko];
            pf.q[0] = *(const uint4*)(pp);
            pf.q[1] = *(const uint4*)(pp + 8);
        }

        // ---- O += P * V  (V transposed: Vt[d][s], K-contiguous B fragments) ----
#pragma unroll
        for (int dt = 0; dt < 8; dt++) {
            FragAB vf;
            int dcol = dt * 16 + (lane & 15);
            int kb   = (lane >> 4) * 16;
            const unsigned short* vp = Vbase + (size_t)dcol * S_ + kt0 + kb;
            vf.q[0] = *(const uint4*)(vp);
            vf.q[1] = *(const uint4*)(vp + 8);
            o[dt] = wmma_bf16(pf, vf, o[dt]);
        }
    };

    const int nfull = (q0 + 1) / 32;             // tiles with kt0+31 <= q0
    const int nkt   = (q0 + 16 + 31) / 32;       // all tiles touching causal region
    for (int kt = 0; kt < nfull; kt++) process_tile(kt * 32, false);
    for (int kt = nfull; kt < nkt; kt++) process_tile(kt * 32, true);

    // ---- epilogue: O /= l, store bf16 to attn_out[b*S+q][h*128+d] ----
#pragma unroll
    for (int r = 0; r < 8; r++) {
        float inv = 1.0f / lstat[r];
        int rowq = q0 + r + ((lane >> 4) * 8);
        unsigned short* dst = AO + (size_t)(b * S_ + rowq) * (NH_ * D_) + h * D_;
#pragma unroll
        for (int dt = 0; dt < 8; dt++)
            dst[dt * 16 + (lane & 15)] = f2bf(o[dt][r] * inv);
    }
}

// =====================================================================
// Host-side launcher
// =====================================================================
extern "C" void kernel_launch(void* const* d_in, const int* in_sizes, int n_in,
                              void* d_out, int out_size, void* d_ws, size_t ws_size,
                              hipStream_t stream) {
    (void)in_sizes; (void)n_in; (void)out_size; (void)ws_size;
    const float*     hidden = (const float*)d_in[0];
    const long long* pos    = (const long long*)d_in[1];   // int64 position_ids
    const float*     w_qkv  = (const float*)d_in[2];
    const float*     w_o    = (const float*)d_in[3];
    const float*     qw     = (const float*)d_in[4];
    const float*     kw     = (const float*)d_in[5];

    const int M = B_ * S_;                                  // 4096
    const size_t nHid  = (size_t)M * H_;                    // 8.39M
    const size_t nWqkv = (size_t)OQKV * H_;                 // 8.39M
    const size_t nWo   = (size_t)H_ * (NH_ * D_);           // 4.19M

    // workspace layout (bf16 halves): ~126 MB total
    unsigned short* qkv = (unsigned short*)d_ws;                         // [4096][4096]
    unsigned short* Qb  = qkv + (size_t)M * OQKV;                        // [B,NH,S,D]
    unsigned short* Kb  = Qb  + (size_t)B_ * NH_  * S_ * D_;             // [B,NKV,S,D]
    unsigned short* Vt  = Kb  + (size_t)B_ * NKV_ * S_ * D_;             // [B,NKV,D,S]
    unsigned short* AO  = Vt  + (size_t)B_ * NKV_ * S_ * D_;             // [B*S, NH*D]
    unsigned short* Ab  = AO  + (size_t)M * NH_ * D_;                    // hidden bf16
    unsigned short* Wqb = Ab  + nHid;                                    // w_qkv bf16
    unsigned short* Wob = Wqb + nWqkv;                                   // w_o bf16

    // 0) one-shot fp32 -> bf16 conversion of reused operands
    cvt_f32_to_bf16<<<dim3((unsigned)(nHid  / 4 / 256)), 256, 0, stream>>>(hidden, Ab,  (int)(nHid  / 4));
    cvt_f32_to_bf16<<<dim3((unsigned)(nWqkv / 4 / 256)), 256, 0, stream>>>(w_qkv,  Wqb, (int)(nWqkv / 4));
    cvt_f32_to_bf16<<<dim3((unsigned)(nWo   / 4 / 256)), 256, 0, stream>>>(w_o,    Wob, (int)(nWo   / 4));

    // 1) QKV projection: qkv = hidden * w_qkv^T  (bf16 out)
    wmma_gemm_nt<false><<<dim3(OQKV / 64, M / 64), 128, 0, stream>>>(
        Ab, Wqb, qkv, M, OQKV, H_);

    // 2) RMSNorm + RoPE + repack (V transposed)
    norm_rope_pack<<<dim3((B_ * S_ * 32) / 4), 128, 0, stream>>>(
        qkv, pos, qw, kw, Qb, Kb, Vt);

    // 3) causal GQA flash attention
    flash_attn<<<dim3(B_ * NH_ * (S_ / 16) / 4), 128, 0, stream>>>(Qb, Kb, Vt, AO);

    // 4) output projection: out = AO * w_o^T  (fp32 out)
    wmma_gemm_nt<true><<<dim3(H_ / 64, M / 64), 128, 0, stream>>>(
        AO, Wob, d_out, M, H_, NH_ * D_);
}
